// CNN3_FPB_3719441679191
// MI455X (gfx1250) — compile-verified, matrix operand order checked by reading, not run on hardware
//
#include <hip/hip_runtime.h>

// ---------------------------------------------------------------------------
// Types
// ---------------------------------------------------------------------------
typedef __bf16 bf16;
typedef __attribute__((ext_vector_type(8)))  __bf16 v8bf;
typedef __attribute__((ext_vector_type(16))) __bf16 v16bf;
typedef __attribute__((ext_vector_type(8)))  float  v8f;

__device__ __forceinline__ bf16 f2bf(float f) {
  unsigned u = __builtin_bit_cast(unsigned, f);
  unsigned r = (u + 0x7FFFu + ((u >> 16) & 1u)) >> 16;   // round-to-nearest-even
  unsigned short h = (unsigned short)r;
  return __builtin_bit_cast(bf16, h);
}
__device__ __forceinline__ float bf2f(bf16 b) {
  unsigned short s = __builtin_bit_cast(unsigned short, b);
  unsigned u = ((unsigned)s) << 16;
  return __builtin_bit_cast(float, u);
}
__device__ __forceinline__ v16bf cat8(v8bf lo, v8bf hi) {
  return __builtin_shufflevector(lo, hi, 0,1,2,3,4,5,6,7,8,9,10,11,12,13,14,15);
}
__device__ __forceinline__ void store_val(float* p, float v) { *p = v; }
__device__ __forceinline__ void store_val(bf16*  p, float v) { *p = f2bf(v); }

#define WMMA_BF16(AF, BF, ACC) \
  __builtin_amdgcn_wmma_f32_16x16x32_bf16(false, (AF), false, (BF), (short)0, (ACC), false, false)

// ---------------------------------------------------------------------------
// Tiled WMMA GEMM:  C[m,n] = act( sum_k A[m,k]*W[n,k] + bias[n] (+rowbias) )
//   Block tile 128x64, BK=32, 256 threads = 8 waves.
//   Each wave: 2x2 register tile of 16x16 WMMA accumulators (4 v8f).
//   Double-buffered LDS + software pipeline: one barrier per k-step, global
//   loads for step k+1 issued before the WMMAs of step k.
//   A rows addressed via: A + (m/RPB)*BSTR + (m%RPB)*RSTR   (conv/pad aware)
//   C rows addressed via: C + (m/ORPB)*OBSTR + (m%ORPB)*ORSTR
//   W is N x K row-major bf16.  M % 128 == 0, N % 64 == 0, K % 32 == 0.
// ---------------------------------------------------------------------------
template<bool RELU, bool ROWBIAS, typename OUT_T>
__global__ __launch_bounds__(256) void gemm_bf16_wmma(
    const bf16* __restrict__ A,
    const bf16* __restrict__ W,
    const float* __restrict__ bias,
    const float* __restrict__ rowbias, int rowbias_ld,
    OUT_T* __restrict__ C,
    int K,
    int RPB,  long long BSTR,  int RSTR,
    int ORPB, long long OBSTR, int ORSTR)
{
  const int t  = threadIdx.x;
  const int m0 = blockIdx.x * 128;
  const int n0 = blockIdx.y * 64;

  __shared__ __align__(16) bf16 sA[2][128 * 32];
  __shared__ __align__(16) bf16 sB[2][64 * 32];

  // Cooperative loaders: A tile 128x32 -> 512 16B chunks (2/thread),
  //                      B tile  64x32 -> 256 16B chunks (1/thread).
  const int ar0 = t >> 2;           // rows 0..63
  const int ar1 = ar0 + 64;         // rows 64..127
  const int ac  = (t & 3) * 8;      // k-offset 0,8,16,24
  const long long gmA0 = (long long)m0 + ar0;
  const long long gmA1 = (long long)m0 + ar1;
  const bf16* Arow0 = A + (gmA0 / RPB) * BSTR + (gmA0 % RPB) * (long long)RSTR + ac;
  const bf16* Arow1 = A + (gmA1 / RPB) * BSTR + (gmA1 % RPB) * (long long)RSTR + ac;
  const bf16* Wrow  = W + (long long)(n0 + ar0) * K + ac;
  const int sOffA0 = ar0 * 32 + ac;
  const int sOffA1 = ar1 * 32 + ac;
  const int sOffB  = ar0 * 32 + ac;

  // WMMA fragment mapping
  const int lane = t & 31;
  const int wv   = t >> 5;            // 0..7
  const int rp   = (wv & 3) * 2;      // row 16-block base (of 8)
  const int cp   = (wv >> 2) * 2;     // col 16-block base (of 4)
  const int row  = lane & 15;
  const int half = lane >> 4;

  v8f acc00 = {}, acc01 = {}, acc10 = {}, acc11 = {};

  const int nk = K >> 5;

  // prologue: stage k-step 0 into buffer 0
  v8bf a0 = *(const v8bf*)(Arow0);
  v8bf a1 = *(const v8bf*)(Arow1);
  v8bf b0 = *(const v8bf*)(Wrow);
  *(v8bf*)&sA[0][sOffA0] = a0;
  *(v8bf*)&sA[0][sOffA1] = a1;
  *(v8bf*)&sB[0][sOffB]  = b0;
  __syncthreads();

  for (int ks = 0; ks < nk; ++ks) {
    const int cur = ks & 1;
    // issue global loads for next step early (hidden behind WMMAs)
    if (ks + 1 < nk) {
      const int k1 = (ks + 1) << 5;
      a0 = *(const v8bf*)(Arow0 + k1);
      a1 = *(const v8bf*)(Arow1 + k1);
      b0 = *(const v8bf*)(Wrow + k1);
    }

    const bf16* ap0 = &sA[cur][(rp * 16 + row) * 32 + half * 8];
    const bf16* ap1 = ap0 + 16 * 32;
    v16bf af0 = cat8(*(const v8bf*)ap0, *(const v8bf*)(ap0 + 16));
    v16bf af1 = cat8(*(const v8bf*)ap1, *(const v8bf*)(ap1 + 16));
    const bf16* bp0 = &sB[cur][(cp * 16 + row) * 32 + half * 8];
    const bf16* bp1 = bp0 + 16 * 32;
    v16bf bf0 = cat8(*(const v8bf*)bp0, *(const v8bf*)(bp0 + 16));
    v16bf bf1 = cat8(*(const v8bf*)bp1, *(const v8bf*)(bp1 + 16));

    acc00 = WMMA_BF16(af0, bf0, acc00);
    acc01 = WMMA_BF16(af0, bf1, acc01);
    acc10 = WMMA_BF16(af1, bf0, acc10);
    acc11 = WMMA_BF16(af1, bf1, acc11);

    if (ks + 1 < nk) {
      const int nxt = cur ^ 1;
      *(v8bf*)&sA[nxt][sOffA0] = a0;
      *(v8bf*)&sA[nxt][sOffA1] = a1;
      *(v8bf*)&sB[nxt][sOffB]  = b0;
    }
    __syncthreads();
  }

  // epilogue: C/D layout -> lane holds (row = v + 8*half, col = lane%16)
  #pragma unroll
  for (int i = 0; i < 2; ++i) {
    #pragma unroll
    for (int j = 0; j < 2; ++j) {
      v8f acc = i ? (j ? acc11 : acc10) : (j ? acc01 : acc00);
      const int gn = n0 + (cp + j) * 16 + row;
      const float bv = bias ? bias[gn] : 0.0f;
      #pragma unroll
      for (int v = 0; v < 8; ++v) {
        const long long gm = (long long)m0 + (rp + i) * 16 + half * 8 + v;
        float val = acc[v] + bv;
        if (ROWBIAS) val += rowbias[(gm / RPB) * (long long)rowbias_ld + gn];
        if (RELU)    val = val > 0.0f ? val : 0.0f;
        OUT_T* op = C + (gm / ORPB) * OBSTR + (gm % ORPB) * (long long)ORSTR + gn;
        store_val(op, val);
      }
    }
  }
}

// ---------------------------------------------------------------------------
// Small helper kernels (convert / repack / pad-zero / fc2)
// ---------------------------------------------------------------------------
__global__ void cvt_f32_bf16(bf16* __restrict__ dst, const float* __restrict__ src,
                             long long n) {
  long long i = blockIdx.x * 256LL + threadIdx.x;
  if (i < n) dst[i] = f2bf(src[i]);
}

__global__ void zero_rows(bf16* __restrict__ base, long long bstride, long long off,
                          int nelem, int nb) {
  long long idx = blockIdx.x * 256LL + threadIdx.x;
  long long total = (long long)nb * nelem;
  if (idx >= total) return;
  long long b = idx / nelem;
  int e = (int)(idx % nelem);
  unsigned short z = 0;
  base[b * bstride + off + e] = __builtin_bit_cast(bf16, z);
}

// Wp (64,64,2) -> Wp0[p*64+i], Wp1[p*64+i]
__global__ void repack_patch(bf16* __restrict__ w0, bf16* __restrict__ w1,
                             const float* __restrict__ Wp) {
  int idx = blockIdx.x * 256 + threadIdx.x;
  if (idx >= 64 * 64) return;
  int p = idx >> 6, i = idx & 63;
  w0[idx] = f2bf(Wp[p * 128 + i * 2 + 0]);
  w1[idx] = f2bf(Wp[p * 128 + i * 2 + 1]);
}

// W (O,I,3) row-major -> Wr[o, t*I + i]
__global__ void repack_conv(bf16* __restrict__ dst, const float* __restrict__ src,
                            int O, int I) {
  int idx = blockIdx.x * 256 + threadIdx.x;
  int n = O * I * 3;
  if (idx >= n) return;
  int o = idx / (I * 3);
  int r = idx % (I * 3);
  int i = r / 3, tt = r % 3;
  dst[o * I * 3 + tt * I + i] = f2bf(src[idx]);
}

// Wfc1 (1024,16640): conv part columns ch*64+pos -> pos*256+ch (position-major)
__global__ void repack_fc1(bf16* __restrict__ dst, const float* __restrict__ src) {
  long long idx = blockIdx.x * 256LL + threadIdx.x;
  if (idx >= 1024LL * 16640LL) return;
  int n = (int)(idx / 16640);
  int k = (int)(idx % 16640);
  int sk;
  if (k < 16384) { int pos = k >> 8; int ch = k & 255; sk = ch * 64 + pos; }
  else           { sk = k; }
  dst[idx] = f2bf(src[(long long)n * 16640 + sk]);
}

// FC2: out[b,n] = dot(z1[b,:], Wfc2[n,:]) + bfc2[n]   (M=2048, N=2, K=1024)
__global__ void fc2_kernel(const bf16* __restrict__ z1, const float* __restrict__ W,
                           const float* __restrict__ bias, float* __restrict__ out) {
  int idx = blockIdx.x * 256 + threadIdx.x;
  if (idx >= 2048 * 2) return;
  int b = idx >> 1, n = idx & 1;
  const bf16*  zr = z1 + (long long)b * 1024;
  const float* wr = W + n * 1024;
  float s = bias[n];
  for (int k = 0; k < 1024; ++k) s += bf2f(zr[k]) * wr[k];
  out[idx] = s;
}

// ---------------------------------------------------------------------------
// Launch
// ---------------------------------------------------------------------------
extern "C" void kernel_launch(void* const* d_in, const int* in_sizes, int n_in,
                              void* d_out, int out_size, void* d_ws, size_t ws_size,
                              hipStream_t stream) {
  (void)in_sizes; (void)n_in; (void)out_size; (void)ws_size;
  const float* x    = (const float*)d_in[0];
  const float* Wp   = (const float*)d_in[1];
  const float* bp   = (const float*)d_in[2];
  const float* W1   = (const float*)d_in[3];
  const float* b1   = (const float*)d_in[4];
  const float* W2   = (const float*)d_in[5];
  const float* b2   = (const float*)d_in[6];
  const float* W3   = (const float*)d_in[7];
  const float* b3   = (const float*)d_in[8];
  const float* Wb1  = (const float*)d_in[9];
  const float* bb1  = (const float*)d_in[10];
  const float* Wb2  = (const float*)d_in[11];
  const float* bb2  = (const float*)d_in[12];
  const float* Wb3  = (const float*)d_in[13];
  const float* bb3  = (const float*)d_in[14];
  const float* Wfc1 = (const float*)d_in[15];
  const float* bfc1 = (const float*)d_in[16];
  const float* Wfc2 = (const float*)d_in[17];
  const float* bfc2 = (const float*)d_in[18];
  float* out = (float*)d_out;

  const long long NB = 2048;
  const long long H0B = 257 * 64;    // 16448  padded batch stride (h0)
  const long long H1B = 257 * 128;   // 32896
  const long long H2B = 129 * 256;   // 33024
  const int BIG = 1 << 30;

  // workspace carve-out
  char* wsp = (char*)d_ws;
  auto alloc = [&](long long bytes) {
    char* p = wsp;
    wsp += (bytes + 255) & ~255LL;
    return p;
  };
  bf16*  xb    = (bf16*)alloc(NB * 16384 * 2);
  bf16*  h0    = (bf16*)alloc(NB * H0B * 2);
  bf16*  h1    = (bf16*)alloc(NB * H1B * 2);
  bf16*  h2    = (bf16*)alloc(NB * H2B * 2);
  bf16*  zb    = (bf16*)alloc(NB * 16640 * 2);
  bf16*  z1    = (bf16*)alloc(NB * 1024 * 2);
  float* u0    = (float*)alloc(NB * 64 * 4);
  bf16*  f1b   = (bf16*)alloc(NB * 64 * 2);
  bf16*  f2b   = (bf16*)alloc(NB * 128 * 2);
  bf16*  Wp0b  = (bf16*)alloc(64 * 64 * 2);
  bf16*  Wp1b  = (bf16*)alloc(64 * 64 * 2);
  bf16*  Wb1b  = (bf16*)alloc(64 * 64 * 2);
  bf16*  Wb2b  = (bf16*)alloc(128 * 64 * 2);
  bf16*  Wb3b  = (bf16*)alloc(256 * 128 * 2);
  bf16*  W1r   = (bf16*)alloc(128 * 192 * 2);
  bf16*  W2r   = (bf16*)alloc(256 * 384 * 2);
  bf16*  W3r   = (bf16*)alloc(256 * 768 * 2);
  bf16*  Wfc1r = (bf16*)alloc(1024LL * 16640 * 2);

  // --- stage 0: conversions / repacks / pad zeroing -------------------------
  cvt_f32_bf16<<<dim3((unsigned)((NB * 16384 + 255) / 256)), 256, 0, stream>>>(xb, x, NB * 16384);
  repack_patch<<<16, 256, 0, stream>>>(Wp0b, Wp1b, Wp);
  cvt_f32_bf16<<<16, 256, 0, stream>>>(Wb1b, Wb1, 64 * 64);
  cvt_f32_bf16<<<32, 256, 0, stream>>>(Wb2b, Wb2, 128 * 64);
  cvt_f32_bf16<<<128, 256, 0, stream>>>(Wb3b, Wb3, 256 * 128);
  repack_conv<<<96, 256, 0, stream>>>(W1r, W1, 128, 64);
  repack_conv<<<384, 256, 0, stream>>>(W2r, W2, 256, 128);
  repack_conv<<<768, 256, 0, stream>>>(W3r, W3, 256, 256);
  repack_fc1<<<(unsigned)((1024LL * 16640 + 255) / 256), 256, 0, stream>>>(Wfc1r, Wfc1);

  zero_rows<<<512,  256, 0, stream>>>(h0, H0B, 0,         64,  (int)NB);
  zero_rows<<<512,  256, 0, stream>>>(h0, H0B, 256 * 64,  64,  (int)NB);
  zero_rows<<<1024, 256, 0, stream>>>(h1, H1B, 0,         128, (int)NB);
  zero_rows<<<1024, 256, 0, stream>>>(h1, H1B, 256 * 128, 128, (int)NB);
  zero_rows<<<2048, 256, 0, stream>>>(h2, H2B, 0,         256, (int)NB);

  // --- u0[b,p] = x0 @ Wp0^T + bp  (f32 out, no relu) ------------------------
  gemm_bf16_wmma<false, false, float><<<dim3(16, 1), 256, 0, stream>>>(
      xb, Wp0b, bp, nullptr, 0, u0, 64,
      BIG, 0, 16384,  BIG, 0, 64);

  // --- branch MLP: f1, f2, f3(->zb[:,16384:]) -------------------------------
  gemm_bf16_wmma<true, false, bf16><<<dim3(16, 1), 256, 0, stream>>>(
      xb, Wb1b, bb1, nullptr, 0, f1b, 64,
      BIG, 0, 16384,  BIG, 0, 64);
  gemm_bf16_wmma<true, false, bf16><<<dim3(16, 2), 256, 0, stream>>>(
      f1b, Wb2b, bb2, nullptr, 0, f2b, 64,
      BIG, 0, 64,     BIG, 0, 128);
  gemm_bf16_wmma<true, false, bf16><<<dim3(16, 4), 256, 0, stream>>>(
      f2b, Wb3b, bb3, nullptr, 0, zb + 16384, 128,
      BIG, 0, 128,    BIG, 0, 16640);

  // --- patch embed: h0[b,1+q,:] = relu(x[b,(q+1)*64..] @ Wp1^T + u0[b,:]) ---
  gemm_bf16_wmma<true, true, bf16><<<dim3(4080, 1), 256, 0, stream>>>(
      xb + 64, Wp1b, nullptr, u0, 64, h0 + 64, 64,
      255, 16384, 64,   255, H0B, 64);

  // --- conv1: K=192 contiguous patch, stride 1 ------------------------------
  gemm_bf16_wmma<true, false, bf16><<<dim3(4080, 2), 256, 0, stream>>>(
      h0, W1r, b1, nullptr, 0, h1 + 128, 192,
      255, H0B, 64,     255, H1B, 128);

  // --- conv2: K=384, stride 2 (RSTR = 2*128) --------------------------------
  gemm_bf16_wmma<true, false, bf16><<<dim3(2048, 4), 256, 0, stream>>>(
      h1, W2r, b2, nullptr, 0, h2 + 256, 384,
      128, H1B, 256,    128, H2B, 256);

  // --- conv3: K=768, stride 2 (RSTR = 2*256) -> zb[:, :16384] ---------------
  gemm_bf16_wmma<true, false, bf16><<<dim3(1024, 4), 256, 0, stream>>>(
      h2, W3r, b3, nullptr, 0, zb, 768,
      64, H2B, 512,     64, 16640, 256);

  // --- FC1: (2048 x 16640) @ (16640 x 1024) ---------------------------------
  gemm_bf16_wmma<true, false, bf16><<<dim3(16, 16), 256, 0, stream>>>(
      zb, Wfc1r, bfc1, nullptr, 0, z1, 16640,
      BIG, 0, 16640,    BIG, 0, 1024);

  // --- FC2 ------------------------------------------------------------------
  fc2_kernel<<<16, 256, 0, stream>>>(z1, Wfc2, bfc2, out);
}
